// PPO_LSTM_Agent_48086453846149
// MI455X (gfx1250) — matrix operand order repaired and checked by reading, not compiled, not run on hardware
//
#include <hip/hip_runtime.h>

// Problem constants (PPO LSTM agent: T=512, B=256, D=64, H=128)
#define T_STEPS 512
#define BATCH   256
#define DDIM    64
#define HDIM    128
#define M_TILE  16    // batch rows per workgroup == WMMA M

// LDS strides (elements); all row strides are 16B multiples.
#define A_STRIDE  200  // bf16, per batch row of [x_t | h_masked]
#define B_STRIDE  136  // bf16, per batch row of h_unmasked (projection input)
#define WH_STRIDE 136  // bf16, per row of W_hid

#define SMEM_A0 0                                    // [16][A_STRIDE] buf 0
#define SMEM_A1 (SMEM_A0 + M_TILE * A_STRIDE * 2)    // 6400: buf 1
#define SMEM_B  (SMEM_A1 + M_TILE * A_STRIDE * 2)    // 12800
#define SMEM_WH (SMEM_B  + M_TILE * B_STRIDE * 2)    // 17152
#define SMEM_D  (SMEM_WH + HDIM * WH_STRIDE * 2)     // 51968  1-done[t]
#define SMEM_D2 (SMEM_D  + 16 * 4)                   // 52032  1-done[t+1]
#define SMEM_BYTES (SMEM_D2 + 16 * 4)                // 52096

typedef __bf16 bf16_t;
typedef bf16_t bfv16 __attribute__((ext_vector_type(16)));
typedef bf16_t bfv8  __attribute__((ext_vector_type(8)));
typedef bf16_t bfv4  __attribute__((ext_vector_type(4)));
typedef float  fv8   __attribute__((ext_vector_type(8)));
typedef float  fv4   __attribute__((ext_vector_type(4)));

union FragBF { bfv16 v; bfv8 h[2]; bf16_t e[16]; };

__device__ __forceinline__ float sigf(float x) {
  return 1.0f / (1.0f + __expf(-x));
}
__device__ __forceinline__ float tanhfast(float x) {
  x = fminf(fmaxf(x, -15.0f), 15.0f);
  float e = __expf(2.0f * x);
  return (e - 1.0f) / (e + 1.0f);
}

extern "C" __global__ void __launch_bounds__(256)
ppo_lstm_fused(const float* __restrict__ x,    const float* __restrict__ done,
               const float* __restrict__ h0,   const float* __restrict__ c0,
               const float* __restrict__ W_ih, const float* __restrict__ b_ih,
               const float* __restrict__ W_hh, const float* __restrict__ b_hh,
               const float* __restrict__ W_hid,const float* __restrict__ b_hid,
               float* __restrict__ out)
{
  extern __shared__ char smem[];
  bf16_t* sA0 = (bf16_t*)(smem + SMEM_A0);   // [16][A_STRIDE] bf16 [x | h*mask], even t
  bf16_t* sA1 = (bf16_t*)(smem + SMEM_A1);   // [16][A_STRIDE] bf16 [x | h*mask], odd t
  bf16_t* sB  = (bf16_t*)(smem + SMEM_B);    // [16][B_STRIDE] bf16 h (unmasked)
  bf16_t* sWh = (bf16_t*)(smem + SMEM_WH);   // [HDIM][WH_STRIDE] bf16 W_hid
  float*  sD  = (float*)(smem + SMEM_D);     // [16] 1-done[t]
  float*  sD2 = (float*)(smem + SMEM_D2);    // [16] 1-done[t+1]

  const int tid  = threadIdx.x;
  const int wv   = tid >> 5;        // wave 0..7
  const int lane = tid & 31;
  const int half = lane >> 4;       // 0: lanes 0-15, 1: lanes 16-31
  const int lm   = lane & 15;
  const int m0   = blockIdx.x * M_TILE;

  // ---- setup: register-resident weight B-fragments ----
  // Wave w owns gate tiles {w, 8+w, 16+w, 24+w}: i,f,g,o for columns [16w,16w+16).
  // B-fragment: lane holds 16 contiguous K for column n.  K layout: [x(0..63) | h(64..191)]
  FragBF wb[6][4];   // [ks][gate]
#pragma unroll
  for (int ks = 0; ks < 6; ++ks) {
#pragma unroll
    for (int tn = 0; tn < 4; ++tn) {
      int n = (8 * tn + wv) * M_TILE + lm;
      const float* src = (ks < 2)
          ? (W_ih + n * DDIM + ks * 32 + half * 16)           // x part, K<64
          : (W_hh + n * HDIM + (ks * 32 - DDIM) + half * 16); // h part
      FragBF u;
#pragma unroll
      for (int i = 0; i < 16; ++i) u.e[i] = (bf16_t)src[i];
      wb[ks][tn] = u;
    }
  }
  // bias per gate column (added post-GEMM, so WMMA C can seed from inline 0)
  float bseed[4];
#pragma unroll
  for (int tn = 0; tn < 4; ++tn) {
    int n = (8 * tn + wv) * M_TILE + lm;
    bseed[tn] = b_ih[n] + b_hh[n];
  }
  const float bhid = b_hid[wv * M_TILE + lm];

  // W_hid -> LDS bf16 (small; re-read per step is cheap)
  for (int idx = tid; idx < HDIM * HDIM; idx += 256) {
    int n = idx >> 7, k = idx & 127;
    sWh[n * WH_STRIDE + k] = (bf16_t)W_hid[n * HDIM + k];
  }
  // initial h (pre-masked with done[0]) -> buffer 0 h-part
  for (int idx = tid; idx < M_TILE * HDIM; idx += 256) {
    int m = idx >> 7, j = idx & 127;
    float mk = 1.0f - done[m0 + m];
    sA0[m * A_STRIDE + DDIM + j] = (bf16_t)(h0[(m0 + m) * HDIM + j] * mk);
  }
  // cell state in registers, C-matrix layout: (r,lane) -> row r+8*half, col 16w+lm
  fv8 creg;
#pragma unroll
  for (int r = 0; r < 8; ++r)
    creg[r] = c0[(size_t)(m0 + r + 8 * half) * HDIM + wv * M_TILE + lm];
  __syncthreads();

  for (int t = 0; t < T_STEPS; ++t) {
    bf16_t* Acur = (t & 1) ? sA1 : sA0;
    bf16_t* Anxt = (t & 1) ? sA0 : sA1;

    // ---- phase 1: x_t -> Acur x-part (packed bf16x4); stage masks ----
    {
      const float* xt = x + ((size_t)t * BATCH + m0) * DDIM;
      int e = tid * 4;                      // 4 consecutive d in one row
      int m = e >> 6, d = e & 63;
      fv4 xv = *(const fv4*)(xt + m * DDIM + d);
      bfv4 xb;
#pragma unroll
      for (int i = 0; i < 4; ++i) xb[i] = (bf16_t)xv[i];
      *(bfv4*)(Acur + m * A_STRIDE + d) = xb;   // one ds_store_b64

      if (tid < 16) {
        sD[tid]  = 1.0f - done[(size_t)t * BATCH + m0 + tid];
        sD2[tid] = (t + 1 < T_STEPS)
                     ? 1.0f - done[(size_t)(t + 1) * BATCH + m0 + tid] : 1.0f;
      }
      if (t + 1 < T_STEPS && tid < 64) {    // prefetch next x slab (4KB)
        const float* xn = x + ((size_t)(t + 1) * BATCH + m0) * DDIM;
        __builtin_prefetch(xn + tid * 16, 0, 0);
      }
    }
    __syncthreads();   // B1: x(t) + masks visible to all waves

    // ---- phase 2: gates = [x|h](16x192) x Wc^T, zero-seeded accumulators ----
    fv8 acc[4] = {};
#pragma unroll
    for (int ks = 0; ks < 6; ++ks) {
      FragBF fa;
      const bf16_t* ar = Acur + lm * A_STRIDE + ks * 32 + half * 16;
      fa.h[0] = *(const bfv8*)(ar);          // K = ks*32 + half*16 + 0..7
      fa.h[1] = *(const bfv8*)(ar + 8);      // K = ks*32 + half*16 + 8..15
#pragma unroll
      for (int tn = 0; tn < 4; ++tn)
        acc[tn] = __builtin_amdgcn_wmma_f32_16x16x32_bf16(
            false, fa.v, false, wb[ks][tn].v, (short)0, acc[tn], false, false);
    }
    // in-register LSTM cell: acc[0]=i, acc[1]=f, acc[2]=g, acc[3]=o (+bias here)
    fv4 md0 = *(const fv4*)(sD + half * 8);
    fv4 md1 = *(const fv4*)(sD + half * 8 + 4);
    fv8 hreg;
#pragma unroll
    for (int r = 0; r < 8; ++r) {
      float mk = (r < 4) ? md0[r] : md1[r - 4];
      float cv = sigf(acc[1][r] + bseed[1]) * (creg[r] * mk)
               + sigf(acc[0][r] + bseed[0]) * tanhfast(acc[2][r] + bseed[2]);
      float hv = sigf(acc[3][r] + bseed[3]) * tanhfast(cv);
      creg[r] = cv;
      hreg[r] = hv;
    }

    // ---- phase 3: publish h: unmasked -> sB (projection), masked(t+1) -> Anxt ----
    {
      fv4 n0 = *(const fv4*)(sD2 + half * 8);
      fv4 n1 = *(const fv4*)(sD2 + half * 8 + 4);
#pragma unroll
      for (int r = 0; r < 8; ++r) {
        int m = r + 8 * half;
        int j = wv * M_TILE + lm;
        sB[m * B_STRIDE + j] = (bf16_t)hreg[r];
        float mk = (r < 4) ? n0[r] : n1[r - 4];
        Anxt[m * A_STRIDE + DDIM + j] = (bf16_t)(hreg[r] * mk);
      }
    }
    __syncthreads();   // B2: sB + Anxt h-part complete

    // ---- phase 4: hidden_t = h @ W_hid^T + b_hid (one 16x16 tile per wave) ----
    {
      fv8 oacc = {};
#pragma unroll
      for (int ks = 0; ks < 4; ++ks) {
        FragBF fa, fb;
        const bf16_t* ar = sB + lm * B_STRIDE + ks * 32 + half * 16;
        fa.h[0] = *(const bfv8*)(ar);
        fa.h[1] = *(const bfv8*)(ar + 8);
        const bf16_t* br = sWh + (wv * M_TILE + lm) * WH_STRIDE + ks * 32 + half * 16;
        fb.h[0] = *(const bfv8*)(br);
        fb.h[1] = *(const bfv8*)(br + 8);
        oacc = __builtin_amdgcn_wmma_f32_16x16x32_bf16(
            false, fa.v, false, fb.v, (short)0, oacc, false, false);
      }
      float* outt = out + ((size_t)t * BATCH + m0) * HDIM;
#pragma unroll
      for (int r = 0; r < 8; ++r)   // C layout: (r,lane) -> row r+8*half, col lm
        outt[(size_t)(r + half * 8) * HDIM + wv * M_TILE + lm] = oacc[r] + bhid;
    }
    // no loop-end barrier: next phase 1 writes the OTHER A buffer's x-part and
    // sD/sD2 (readers of which finished before B2); sB readers finish before
    // the next B1; ordering holds with 2 barriers per step.
  }
}

extern "C" void kernel_launch(void* const* d_in, const int* in_sizes, int n_in,
                              void* d_out, int out_size, void* d_ws, size_t ws_size,
                              hipStream_t stream) {
  (void)in_sizes; (void)n_in; (void)d_ws; (void)ws_size; (void)out_size;
  const float* x     = (const float*)d_in[0];
  const float* done  = (const float*)d_in[1];
  const float* h0    = (const float*)d_in[2];
  const float* c0    = (const float*)d_in[3];
  const float* W_ih  = (const float*)d_in[4];
  const float* b_ih  = (const float*)d_in[5];
  const float* W_hh  = (const float*)d_in[6];
  const float* b_hh  = (const float*)d_in[7];
  const float* W_hid = (const float*)d_in[8];
  const float* b_hid = (const float*)d_in[9];
  float* out = (float*)d_out;

  (void)hipFuncSetAttribute((const void*)ppo_lstm_fused,
                            hipFuncAttributeMaxDynamicSharedMemorySize,
                            SMEM_BYTES);

  dim3 grid(BATCH / M_TILE);   // 16 workgroups, one per 16-row batch slice
  dim3 block(256);             // 8 wave32s
  ppo_lstm_fused<<<grid, block, SMEM_BYTES, stream>>>(
      x, done, h0, c0, W_ih, b_ih, W_hh, b_hh, W_hid, b_hid, out);
}
